// ResidualBlock_20753281974541
// MI455X (gfx1250) — matrix operand order repaired and checked by reading, not compiled, not run on hardware
//
#include <hip/hip_runtime.h>
#include <math.h>

typedef __attribute__((ext_vector_type(16))) __bf16 v16bf;
typedef __attribute__((ext_vector_type(8)))  __bf16 v8bf;
typedef __attribute__((ext_vector_type(8)))  float  v8f;
typedef __attribute__((ext_vector_type(4)))  unsigned int u32x4;
typedef __attribute__((ext_vector_type(8)))  int i32x8;
typedef __attribute__((ext_vector_type(4)))  int i32x4;

#define B_SZ    4
#define L_SEQ   1024
#define DMODEL  1024
#define DINNER  2048
#define NSTATE  16
#define DTRANK  64
#define KCONV   4
#define NTOK    (B_SZ * L_SEQ)          // 4096 tokens
#define PROJ_N  (2 * DINNER)            // 4096
#define XPAD    128                     // padded x_dbl width (real width 96)

__device__ __forceinline__ float silu_f(float v) { return v / (1.f + __expf(-v)); }

// ---------------------------------------------------------------- TDM tile load (A tile)
// 2D descriptor: tile = 32 (K, contiguous) x 128 (M rows), bf16 elements, row stride = K.
// LDS padding: 4 DWORDs inserted every 16 DWORDs -> LDS row stride = 40 bf16 = sA[TK+8] layout.
__device__ __forceinline__ void tdm_load_tile_a(const __bf16* gptr, unsigned lds_off, int K) {
  const unsigned long long ga = (unsigned long long)(uintptr_t)gptr;
  u32x4 g0;
  g0[0] = 1u;                                            // count=1, user descriptor
  g0[1] = lds_off;                                       // lds_addr (bytes)
  g0[2] = (unsigned)(ga & 0xffffffffu);                  // global_addr[31:0]
  g0[3] = (unsigned)((ga >> 32) & 0x1ffffffu) | (2u << 30); // global_addr[56:32] | type=2
  i32x8 g1;
  g1[0] = (int)((1u << 16) |                             // data_size = 1 -> 2-byte elems
                (1u << 20) |                             // pad_enable
                (3u << 22) |                             // pad_interval: 16 DWORDs
                (3u << 25));                             // pad_amount: 4 DWORDs
  g1[1] = (int)(32u << 16);                              // tensor_dim0 = 32
  g1[2] = (int)(128u << 16);                             // tensor_dim1 = 128
  g1[3] = (int)(32u << 16);                              // tile_dim0 = 32
  g1[4] = 128;                                           // tile_dim1 = 128 (tile_dim2 = 0)
  g1[5] = K;                                             // tensor_dim0_stride = K elements
  g1[6] = 0;
  g1[7] = 0;
  i32x4 z4 = {0, 0, 0, 0};
#if defined(__clang_major__) && (__clang_major__ >= 23)
  i32x8 z8 = {0, 0, 0, 0, 0, 0, 0, 0};
  __builtin_amdgcn_tensor_load_to_lds(g0, g1, z4, z4, z8, 0);
#else
  __builtin_amdgcn_tensor_load_to_lds(g0, g1, z4, z4, 0);
#endif
}

// ---------------------------------------------------------------- elementwise convert
__global__ void cvt_bf16_kernel(const float* __restrict__ src, __bf16* __restrict__ dst, int n) {
  int i = blockIdx.x * blockDim.x + threadIdx.x;
  if (i < n) dst[i] = (__bf16)src[i];
}

// convert with zero-padding of trailing columns (rows x cols -> rows x cols_pad)
__global__ void cvt_bf16_pad_kernel(const float* __restrict__ src, __bf16* __restrict__ dst,
                                    int rows, int cols, int cols_pad) {
  int i = blockIdx.x * blockDim.x + threadIdx.x;
  if (i >= rows * cols_pad) return;
  int r = i / cols_pad, c = i % cols_pad;
  dst[i] = (c < cols) ? (__bf16)src[(size_t)r * cols + c] : (__bf16)0.f;
}

// ---------------------------------------------------------------- layernorm -> bf16
__global__ __launch_bounds__(256) void layernorm_kernel(
    const float* __restrict__ x, const float* __restrict__ g,
    const float* __restrict__ b, __bf16* __restrict__ out) {
  __shared__ float s1[256], s2[256];
  const int row = blockIdx.x;
  const float* xr = x + (size_t)row * DMODEL;
  float sum = 0.f, sq = 0.f;
  for (int c = threadIdx.x; c < DMODEL; c += 256) { float v = xr[c]; sum += v; sq += v * v; }
  s1[threadIdx.x] = sum; s2[threadIdx.x] = sq; __syncthreads();
  for (int s = 128; s > 0; s >>= 1) {
    if (threadIdx.x < s) { s1[threadIdx.x] += s1[threadIdx.x + s]; s2[threadIdx.x] += s2[threadIdx.x + s]; }
    __syncthreads();
  }
  const float mu  = s1[0] / DMODEL;
  const float var = s2[0] / DMODEL - mu * mu;
  const float rs  = rsqrtf(var + 1e-5f);
  for (int c = threadIdx.x; c < DMODEL; c += 256) {
    float v = (xr[c] - mu) * rs * g[c] + b[c];
    out[(size_t)row * DMODEL + c] = (__bf16)v;
  }
}

// ---------------------------------------------------------------- bf16 WMMA GEMM (exact tiles)
// Requires M % 128 == 0, N % 64 == 0, K % 32 == 0 (guaranteed by launcher).
// A tile staged by the Tensor Data Mover (wave 0 issues, TENSORcnt-synced);
// B tile staged via register double-buffer + transposed LDS scatter.
// C[M,N](f32) = act( A[M,K](bf16) @ B[K,N](bf16) + bias ) + resid ; act: 0=none, 1=softplus
#define TM 128
#define TN 64
#define TK 32
#define LDP 8

__global__ __launch_bounds__(256) void wmma_gemm_bf16_kernel(
    const __bf16* __restrict__ A, const __bf16* __restrict__ Bm,
    const float* __restrict__ bias, const float* __restrict__ resid,
    float* __restrict__ C, int M, int N, int K, int act) {
  __shared__ __bf16 sA[2][TM][TK + LDP];   // 20.0 KB, filled by TDM
  __shared__ __bf16 sBt[2][TN][TK + LDP];  // 10.0 KB (B tile transposed: n-major)

  const int tid   = threadIdx.x;
  const int wave  = tid >> 5;
  const int lane  = tid & 31;
  const int lhalf = lane >> 4;
  const int l16   = lane & 15;
  const int wm    = wave >> 1;   // 0..3 : 32-row wave subtile
  const int wn    = wave & 1;    // 0..1 : 32-col wave subtile
  const int m0    = blockIdx.y * TM;
  const int n0    = blockIdx.x * TN;

  // B staging indices: 32x64 tile, 8 elems/thread
  const int bk = tid >> 3;            // 0..31
  const int bn = (tid & 7) << 3;      // 0..56
  const __bf16* bptr  = Bm + (size_t)bk * N + (n0 + bn);
  const __bf16* aTile = A + (size_t)m0 * K;     // advance by t*TK columns

  const unsigned ldsA0 = (unsigned)(uintptr_t)(&sA[0][0][0]);
  const unsigned ldsA1 = (unsigned)(uintptr_t)(&sA[1][0][0]);

  // ---- preload tile 0
  if (wave == 0) tdm_load_tile_a(aTile, ldsA0, K);
  v8bf b0 = *(const v8bf*)(bptr);
#pragma unroll
  for (int i = 0; i < 8; ++i) sBt[0][bn + i][bk] = b0[i];
  if (wave == 0) __builtin_amdgcn_s_wait_tensorcnt(0);
  __syncthreads();

  v8f acc00 = {}, acc01 = {}, acc10 = {}, acc11 = {};

  const int nk = K / TK;
  for (int t = 0; t < nk; ++t) {
    const int buf = t & 1;
    const bool more = (t + 1) < nk;
    if (more) {  // fetch next tile: TDM for A, registers for B (overlaps WMMAs below)
      if (wave == 0) tdm_load_tile_a(aTile + (size_t)(t + 1) * TK, buf ? ldsA0 : ldsA1, K);
      b0 = *(const v8bf*)(bptr + (size_t)(t + 1) * TK * N);
    }
    // A fragments (16x32): lane<16 holds K[0..7],K[16..23]; lane>=16 holds K[8..15],K[24..31]
    v16bf af0, af1, bfr0, bfr1;
    {
      const __bf16* r0 = &sA[buf][wm * 32 + l16][0];
      const __bf16* r1 = &sA[buf][wm * 32 + 16 + l16][0];
#pragma unroll
      for (int i = 0; i < 8; ++i) {
        af0[i] = r0[lhalf * 8 + i]; af0[8 + i] = r0[16 + lhalf * 8 + i];
        af1[i] = r1[lhalf * 8 + i]; af1[8 + i] = r1[16 + lhalf * 8 + i];
      }
      // B fragments (32x16): lane<16 holds K[0..7],K[16..23]; lane>=16 holds K[8..15],K[24..31]
      const __bf16* c0 = &sBt[buf][wn * 32 + l16][0];
      const __bf16* c1 = &sBt[buf][wn * 32 + 16 + l16][0];
#pragma unroll
      for (int i = 0; i < 8; ++i) {
        bfr0[i] = c0[lhalf * 8 + i]; bfr0[8 + i] = c0[16 + lhalf * 8 + i];
        bfr1[i] = c1[lhalf * 8 + i]; bfr1[8 + i] = c1[16 + lhalf * 8 + i];
      }
    }
    acc00 = __builtin_amdgcn_wmma_f32_16x16x32_bf16(false, af0, false, bfr0, (short)0, acc00, false, false);
    acc01 = __builtin_amdgcn_wmma_f32_16x16x32_bf16(false, af0, false, bfr1, (short)0, acc01, false, false);
    acc10 = __builtin_amdgcn_wmma_f32_16x16x32_bf16(false, af1, false, bfr0, (short)0, acc10, false, false);
    acc11 = __builtin_amdgcn_wmma_f32_16x16x32_bf16(false, af1, false, bfr1, (short)0, acc11, false, false);
    if (more) {  // stage next B tile into the other LDS buffer
      const int nb = buf ^ 1;
#pragma unroll
      for (int i = 0; i < 8; ++i) sBt[nb][bn + i][bk] = b0[i];
    }
    if (wave == 0) __builtin_amdgcn_s_wait_tensorcnt(0);  // tile t+1 TDM done
    __syncthreads();
  }

  // epilogue: C/D layout -> VGPR i: row lhalf*8+i, col l16
#pragma unroll
  for (int i = 0; i < 8; ++i) {
    const int gm0 = m0 + wm * 32 + lhalf * 8 + i;
    const int gm1 = gm0 + 16;
    const int gn0 = n0 + wn * 32 + l16;
    const int gn1 = gn0 + 16;
    float v00 = acc00[i], v01 = acc01[i], v10 = acc10[i], v11 = acc11[i];
    if (bias) {
      const float bb0 = bias[gn0], bb1 = bias[gn1];
      v00 += bb0; v10 += bb0; v01 += bb1; v11 += bb1;
    }
    if (act == 1) {
      v00 = (v00 > 20.f) ? v00 : log1pf(__expf(v00));
      v01 = (v01 > 20.f) ? v01 : log1pf(__expf(v01));
      v10 = (v10 > 20.f) ? v10 : log1pf(__expf(v10));
      v11 = (v11 > 20.f) ? v11 : log1pf(__expf(v11));
    }
    if (resid) {
      v00 += resid[(size_t)gm0 * N + gn0]; v01 += resid[(size_t)gm0 * N + gn1];
      v10 += resid[(size_t)gm1 * N + gn0]; v11 += resid[(size_t)gm1 * N + gn1];
    }
    C[(size_t)gm0 * N + gn0] = v00; C[(size_t)gm0 * N + gn1] = v01;
    C[(size_t)gm1 * N + gn0] = v10; C[(size_t)gm1 * N + gn1] = v11;
  }
}

// ---------------------------------------------------------------- causal depthwise conv + SiLU
__global__ void conv_silu_kernel(const float* __restrict__ proj,
                                 const float* __restrict__ cw, const float* __restrict__ cb,
                                 float* __restrict__ uf, __bf16* __restrict__ ub) {
  const int idx = blockIdx.x * blockDim.x + threadIdx.x;
  if (idx >= B_SZ * L_SEQ * DINNER) return;
  const int d = idx % DINNER;
  const int l = (idx / DINNER) % L_SEQ;
  const int b = idx / (DINNER * L_SEQ);
  float acc = cb[d];
#pragma unroll
  for (int i = 0; i < KCONV; ++i) {
    const int ls = l - (KCONV - 1) + i;
    if (ls >= 0) acc += cw[i * DINNER + d] * proj[((size_t)(b * L_SEQ + ls)) * PROJ_N + d];
  }
  const float v = silu_f(acc);
  uf[idx] = v;
  ub[idx] = (__bf16)v;
}

// ---------------------------------------------------------------- extract dt columns -> bf16
__global__ void extract_dt_kernel(const float* __restrict__ xdbl, __bf16* __restrict__ dt) {
  const int i = blockIdx.x * blockDim.x + threadIdx.x;
  if (i >= NTOK * DTRANK) return;
  const int m = i / DTRANK, j = i % DTRANK;
  dt[i] = (__bf16)xdbl[(size_t)m * XPAD + j];
}

// ---------------------------------------------------------------- selective scan + fused gate
#define SCHUNK 64
__global__ __launch_bounds__(256) void scan_kernel(
    const float* __restrict__ u, const float* __restrict__ delta,
    const float* __restrict__ xdbl, const float* __restrict__ A_log,
    const float* __restrict__ Dv, const float* __restrict__ proj,
    __bf16* __restrict__ yg) {
  __shared__ float sBC[SCHUNK][2 * NSTATE];  // [step][0..15]=B, [16..31]=C
  const int tid = threadIdx.x;
  const int b = blockIdx.x / (DINNER / 256);
  const int d = (blockIdx.x % (DINNER / 256)) * 256 + tid;

  float A[NSTATE], h[NSTATE];
#pragma unroll
  for (int n = 0; n < NSTATE; ++n) { A[n] = -__expf(A_log[(size_t)d * NSTATE + n]); h[n] = 0.f; }
  const float Dval = Dv[d];

  for (int l0 = 0; l0 < L_SEQ; l0 += SCHUNK) {
    __syncthreads();
    for (int q = tid; q < SCHUNK * 2 * NSTATE; q += 256) {
      const int st = q >> 5, c = q & 31;   // xdbl cols 64..95 == [B | C]
      sBC[st][c] = xdbl[((size_t)(b * L_SEQ) + l0 + st) * XPAD + DTRANK + c];
    }
    __syncthreads();
    for (int s = 0; s < SCHUNK; ++s) {
      const size_t row = (size_t)(b * L_SEQ) + l0 + s;
      const size_t off = row * DINNER + d;
      const float dt = delta[off];
      const float ut = u[off];
      const float du = dt * ut;
      float acc = 0.f;
#pragma unroll
      for (int n = 0; n < NSTATE; ++n) {
        const float dA = __expf(dt * A[n]);
        const float hn = dA * h[n] + du * sBC[s][n];
        h[n] = hn;
        acc += hn * sBC[s][NSTATE + n];
      }
      const float yv = acc + ut * Dval;
      const float r  = proj[row * PROJ_N + DINNER + d];   // res half of in-projection
      yg[off] = (__bf16)(yv * silu_f(r));                 // fused output gate
    }
  }
}

// ---------------------------------------------------------------- launcher
extern "C" void kernel_launch(void* const* d_in, const int* in_sizes, int n_in,
                              void* d_out, int out_size, void* d_ws, size_t ws_size,
                              hipStream_t stream) {
  (void)in_sizes; (void)n_in; (void)out_size; (void)ws_size;
  const float* x      = (const float*)d_in[0];
  const float* ln_g   = (const float*)d_in[1];
  const float* ln_b   = (const float*)d_in[2];
  const float* W_in   = (const float*)d_in[3];
  const float* b_in   = (const float*)d_in[4];
  const float* conv_w = (const float*)d_in[5];
  const float* conv_b = (const float*)d_in[6];
  const float* W_x    = (const float*)d_in[7];
  const float* W_dt   = (const float*)d_in[8];
  const float* b_dt   = (const float*)d_in[9];
  const float* A_log  = (const float*)d_in[10];
  const float* Dvec   = (const float*)d_in[11];
  const float* W_out  = (const float*)d_in[12];
  const float* b_out  = (const float*)d_in[13];
  float* out = (float*)d_out;

  char* ws = (char*)d_ws;
  size_t off = 0;
  auto alloc = [&](size_t bytes) -> char* {
    char* p = ws + off; off += (bytes + 255) & ~(size_t)255; return p;
  };
  __bf16* Win_bf  = (__bf16*)alloc((size_t)DMODEL * PROJ_N * 2);
  __bf16* Wout_bf = (__bf16*)alloc((size_t)DINNER * DMODEL * 2);
  __bf16* Wx_bf   = (__bf16*)alloc((size_t)DINNER * XPAD * 2);     // padded to 128 cols
  __bf16* Wdt_bf  = (__bf16*)alloc((size_t)DTRANK * DINNER * 2);
  __bf16* xn_bf   = (__bf16*)alloc((size_t)NTOK * DMODEL * 2);
  float*  proj    = (float*) alloc((size_t)NTOK * PROJ_N * 4);
  float*  u_f     = (float*) alloc((size_t)NTOK * DINNER * 4);
  __bf16* u_bf    = (__bf16*)alloc((size_t)NTOK * DINNER * 2);
  float*  xdbl    = (float*) alloc((size_t)NTOK * XPAD * 4);       // padded to 128 cols
  __bf16* dt_bf   = (__bf16*)alloc((size_t)NTOK * DTRANK * 2);
  float*  delta   = (float*) alloc((size_t)NTOK * DINNER * 4);
  __bf16* yg      = (__bf16*)alloc((size_t)NTOK * DINNER * 2);

  auto g1 = [](int n) { return dim3((unsigned)((n + 255) / 256)); };
  const dim3 blk(256);

  // weight conversions
  cvt_bf16_kernel<<<g1(DMODEL * PROJ_N), blk, 0, stream>>>(W_in,  Win_bf,  DMODEL * PROJ_N);
  cvt_bf16_kernel<<<g1(DINNER * DMODEL), blk, 0, stream>>>(W_out, Wout_bf, DINNER * DMODEL);
  cvt_bf16_pad_kernel<<<g1(DINNER * XPAD), blk, 0, stream>>>(W_x, Wx_bf, DINNER, DTRANK + 2 * NSTATE, XPAD);
  cvt_bf16_kernel<<<g1(DTRANK * DINNER), blk, 0, stream>>>(W_dt,  Wdt_bf,  DTRANK * DINNER);

  // 1) layernorm
  layernorm_kernel<<<dim3(NTOK), blk, 0, stream>>>(x, ln_g, ln_b, xn_bf);
  // 2) in-projection: proj = xn @ W_in + b_in   [4096 x 4096], K=1024
  wmma_gemm_bf16_kernel<<<dim3(PROJ_N / TN, NTOK / TM), blk, 0, stream>>>(
      xn_bf, Win_bf, b_in, nullptr, proj, NTOK, PROJ_N, DMODEL, 0);
  // 3) depthwise causal conv + SiLU on u-half of proj
  conv_silu_kernel<<<g1(NTOK * DINNER), blk, 0, stream>>>(proj, conv_w, conv_b, u_f, u_bf);
  // 4) x_dbl = u @ W_x (padded)   [4096 x 128], K=2048
  wmma_gemm_bf16_kernel<<<dim3(XPAD / TN, NTOK / TM), blk, 0, stream>>>(
      u_bf, Wx_bf, nullptr, nullptr, xdbl, NTOK, XPAD, DINNER, 0);
  // 5) delta = softplus(dt @ W_dt + b_dt)   [4096 x 2048], K=64
  extract_dt_kernel<<<g1(NTOK * DTRANK), blk, 0, stream>>>(xdbl, dt_bf);
  wmma_gemm_bf16_kernel<<<dim3(DINNER / TN, NTOK / TM), blk, 0, stream>>>(
      dt_bf, Wdt_bf, b_dt, nullptr, delta, NTOK, DINNER, DTRANK, 1);
  // 6) selective scan with fused output gate -> bf16
  scan_kernel<<<dim3(B_SZ * (DINNER / 256)), blk, 0, stream>>>(
      u_f, delta, xdbl, A_log, Dvec, proj, yg);
  // 7) out-projection + bias + residual: out = yg @ W_out + b_out + x
  wmma_gemm_bf16_kernel<<<dim3(DMODEL / TN, NTOK / TM), blk, 0, stream>>>(
      yg, Wout_bf, b_out, x, out, NTOK, DMODEL, DINNER, 0);
}
